// MultiheadAttention_64269890617996
// MI455X (gfx1250) — matrix operand order repaired and checked by reading, not compiled, not run on hardware
//
#include <hip/hip_runtime.h>
#include <hip/hip_bf16.h>

// ---------------------------------------------------------------------------
// MultiheadAttention forward for MI455X (gfx1250, wave32, WMMA).
// B=4, T=2048, E=1024, H=16, D=64.
// bf16 WMMA (16x16x32, f32 accum) for all matrix stages, flash-attention
// online softmax (DPP butterflies), double-buffered async global->LDS staging
// of K/V chunks (ASYNCcnt + workgroup barrier).
// ---------------------------------------------------------------------------

#define BATCH 4
#define SEQ   2048
#define EMB   1024
#define HEADS 16
#define HDIM  64
#define BH    (BATCH * HEADS)   // 64
#define MTOT  (BATCH * SEQ)     // 8192

typedef __attribute__((ext_vector_type(16))) __bf16 bf16x16;
typedef __attribute__((ext_vector_type(8)))  __bf16 bf16x8;
typedef __attribute__((ext_vector_type(8)))  float  f32x8;

union FragAB { bf16x16 v; bf16x8 h[2]; };

__device__ __forceinline__ unsigned short f2bf(float x) {
    union { float f; unsigned u; } c; c.f = x;
    unsigned r = (c.u + 0x7FFFu + ((c.u >> 16) & 1u)) >> 16;
    return (unsigned short)r;
}

__device__ __forceinline__ f32x8 wmma_bf16(bf16x16 a, bf16x16 b, f32x8 c) {
    return __builtin_amdgcn_wmma_f32_16x16x32_bf16(
        /*neg_a=*/false, a, /*neg_b=*/false, b,
        /*c_mod=*/(short)0, c, /*reuse_a=*/false, /*reuse_b=*/false);
}

// A-fragment (16x32 bf16, M rows, K cols, memory contiguous along K).
// lane<16: row (lane&15), K-chunks [k,k+8) and [k+16,k+24);
// lane>=16: [k+8,k+16) and [k+24,k+32).
__device__ __forceinline__ bf16x16 load_frag_A(const unsigned short* base,
                                               int stride, int rowIdx, int k, int lane) {
    FragAB f;
    const unsigned short* p = base + (size_t)rowIdx * stride + k + ((lane >> 4) << 3);
    f.h[0] = *reinterpret_cast<const bf16x8*>(p);
    f.h[1] = *reinterpret_cast<const bf16x8*>(p + 16);
    return f.v;
}

// B-fragment (32x16 bf16, K rows, N cols, memory contiguous along K per column).
// lane n(<16): column n, K=[k,k+8) and [k+16,k+24); lane n+16: column n, K=[k+8,k+16) and [k+24,k+32).
__device__ __forceinline__ bf16x16 load_frag_B(const unsigned short* base,
                                               int stride, int colIdx, int k, int lane) {
    FragAB f;
    const unsigned short* p = base + (size_t)colIdx * stride + k + ((lane >> 4) << 3);
    f.h[0] = *reinterpret_cast<const bf16x8*>(p);
    f.h[1] = *reinterpret_cast<const bf16x8*>(p + 16);
    return f.v;
}

// ---- 16-lane butterfly reductions via DPP16 (VALU-only, no LDS pipe) ------
// quad_perm(1,0,3,2)=0xB1, quad_perm(2,3,0,1)=0x4E,
// ROW_HALF_MIRROR=0x141, ROW_MIRROR=0x140.
__device__ __forceinline__ float red16_max(float x) {
    union { float f; int i; } a, b;
    a.f = x;
    b.i = __builtin_amdgcn_mov_dpp(a.i, 0xB1,  0xF, 0xF, true); a.f = fmaxf(a.f, b.f);
    b.i = __builtin_amdgcn_mov_dpp(a.i, 0x4E,  0xF, 0xF, true); a.f = fmaxf(a.f, b.f);
    b.i = __builtin_amdgcn_mov_dpp(a.i, 0x141, 0xF, 0xF, true); a.f = fmaxf(a.f, b.f);
    b.i = __builtin_amdgcn_mov_dpp(a.i, 0x140, 0xF, 0xF, true); a.f = fmaxf(a.f, b.f);
    return a.f;
}
__device__ __forceinline__ float red16_sum(float x) {
    union { float f; int i; } a, b;
    a.f = x;
    b.i = __builtin_amdgcn_mov_dpp(a.i, 0xB1,  0xF, 0xF, true); a.f += b.f;
    b.i = __builtin_amdgcn_mov_dpp(a.i, 0x4E,  0xF, 0xF, true); a.f += b.f;
    b.i = __builtin_amdgcn_mov_dpp(a.i, 0x141, 0xF, 0xF, true); a.f += b.f;
    b.i = __builtin_amdgcn_mov_dpp(a.i, 0x140, 0xF, 0xF, true); a.f += b.f;
    return a.f;
}

// ---- CDNA5 async global->LDS copy (ASYNCcnt) ------------------------------
__device__ __forceinline__ void async_copy_b128(unsigned lds_byte_off, const void* gptr) {
    asm volatile("global_load_async_to_lds_b128 %0, %1, off"
                 :: "v"(lds_byte_off), "v"(gptr)
                 : "memory");
}
__device__ __forceinline__ void wait_async0() {
    asm volatile("s_wait_asynccnt 0" ::: "memory");
}

// ---------------------------------------------------------------------------
// Kernel: f32 -> bf16 conversion
// ---------------------------------------------------------------------------
__global__ void cvt_f32_bf16(const float* __restrict__ src,
                             unsigned short* __restrict__ dst, int n) {
    int i = blockIdx.x * blockDim.x + threadIdx.x;
    if (i < n) dst[i] = f2bf(src[i]);
}

// ---------------------------------------------------------------------------
// Tiled bf16 GEMM: C[M,N] = A[M,K] * W[N,K]^T + bias
// Block tile 128x128, 8 waves (4 in M x 2 in N), wave tile 32x64.
// MODE 0: QKV epilogue (scatter to q/k/v attention layouts, bf16)
// MODE 1: plain f32 output (out projection)
// ---------------------------------------------------------------------------
template <int MODE>
__global__ __launch_bounds__(256) void gemm_bf16_kernel(
    const unsigned short* __restrict__ A,   // [M,K] bf16
    const unsigned short* __restrict__ W,   // [N,K] bf16
    const float* __restrict__ bias,         // [N]
    int M, int N, int K,
    unsigned short* __restrict__ q_out,     // [BH,T,D]  (MODE 0)
    unsigned short* __restrict__ k_out,     // [BH,T,D]  (MODE 0)
    unsigned short* __restrict__ v_out,     // [BH,D,T]  (MODE 0)
    float* __restrict__ out)                // [M,N]     (MODE 1)
{
    const int lane  = threadIdx.x & 31;
    const int wave  = threadIdx.x >> 5;
    const int waveM = wave & 3;
    const int waveN = wave >> 2;
    const int m0 = blockIdx.y * 128 + waveM * 32;
    const int n0 = blockIdx.x * 128 + waveN * 64;
    const int row = lane & 15;

    f32x8 acc[2][4];
#pragma unroll
    for (int i = 0; i < 2; ++i)
#pragma unroll
        for (int j = 0; j < 4; ++j)
#pragma unroll
            for (int r = 0; r < 8; ++r) acc[i][j][r] = 0.0f;

    for (int k = 0; k < K; k += 32) {
        bf16x16 a[2], b[4];
#pragma unroll
        for (int mt = 0; mt < 2; ++mt)
            a[mt] = load_frag_A(A, K, m0 + mt * 16 + row, k, lane);
#pragma unroll
        for (int nt = 0; nt < 4; ++nt)
            b[nt] = load_frag_B(W, K, n0 + nt * 16 + row, k, lane);
        __builtin_prefetch(A + (size_t)(m0 + row) * K + k + 32, 0, 0);
        __builtin_prefetch(W + (size_t)(n0 + row) * K + k + 32, 0, 0);
#pragma unroll
        for (int mt = 0; mt < 2; ++mt)
#pragma unroll
            for (int nt = 0; nt < 4; ++nt)
                acc[mt][nt] = wmma_bf16(a[mt], b[nt], acc[mt][nt]);
    }

    // Epilogue. C layout: VGPR r -> row tile+r+8*(lane>=16), col tile+(lane&15)
#pragma unroll
    for (int mt = 0; mt < 2; ++mt) {
#pragma unroll
        for (int nt = 0; nt < 4; ++nt) {
#pragma unroll
            for (int r = 0; r < 8; ++r) {
                int m = m0 + mt * 16 + r + ((lane >> 4) << 3);
                int n = n0 + nt * 16 + (lane & 15);
                float v = acc[mt][nt][r] + bias[n];
                if (MODE == 0) {
                    int which = n >> 10;          // 0:Q 1:K 2:V
                    int e = n & (EMB - 1);
                    int h = e >> 6, d = e & 63;
                    int bb = m >> 11, t = m & (SEQ - 1);
                    int bh = bb * HEADS + h;
                    if (which == 0)
                        q_out[((size_t)bh * SEQ + t) * HDIM + d] = f2bf(v * 0.125f); // D^-0.5
                    else if (which == 1)
                        k_out[((size_t)bh * SEQ + t) * HDIM + d] = f2bf(v);
                    else
                        v_out[((size_t)bh * HDIM + d) * SEQ + t] = f2bf(v);
                } else {
                    out[(size_t)m * N + n] = v;
                }
            }
        }
    }
}

// ---------------------------------------------------------------------------
// Flash attention: one block per (128 query rows, bh). Each wave owns 16 rows.
// K/V chunks (32 keys) are double-buffered into LDS with async copies shared
// by all 8 waves; online softmax over 2048 keys.
// ---------------------------------------------------------------------------
__global__ __launch_bounds__(256) void flash_attn_kernel(
    const unsigned short* __restrict__ Q,   // [BH,T,D] (pre-scaled)
    const unsigned short* __restrict__ Kt,  // [BH,T,D]
    const unsigned short* __restrict__ Vt,  // [BH,D,T]
    unsigned short* __restrict__ ctx)       // [B*T, E] bf16
{
    __shared__ unsigned short lds_k[2][32 * HDIM];  // 2 x 4 KB
    __shared__ unsigned short lds_v[2][HDIM * 32];  // 2 x 4 KB
    __shared__ unsigned short lds_p[8][16 * 32];    // 8 KB per-wave P staging

    const int tid  = threadIdx.x;
    const int lane = tid & 31;
    const int wave = tid >> 5;
    const int bh = blockIdx.y;
    const int b = bh >> 4, h = bh & 15;
    const int qbase = blockIdx.x * 128 + wave * 16;
    const int row = lane & 15;

    const unsigned short* Qb = Q  + (size_t)bh * SEQ * HDIM;
    const unsigned short* Kb = Kt + (size_t)bh * SEQ * HDIM;
    const unsigned short* Vb = Vt + (size_t)bh * HDIM * SEQ;

    const unsigned kbase = (unsigned)(uintptr_t)&lds_k[0][0];
    const unsigned vbase = (unsigned)(uintptr_t)&lds_v[0][0];

    // cooperative async stage of one 32-key chunk (K: 32x64, V: 64x32) into buf
    auto stage = [&](int buf, int kc) {
        {   // K chunk: 32 rows x 128B; each thread copies 16B
            int r = tid >> 3, s = (tid & 7) * 8;
            const unsigned short* g = Kb + (size_t)(kc + r) * HDIM + s;
            unsigned l = kbase + (unsigned)(buf * (32 * HDIM) + r * HDIM + s) * 2u;
            async_copy_b128(l, g);
        }
        {   // V chunk: 64 rows x 64B; each thread copies 16B
            int d = tid >> 2, s = (tid & 3) * 8;
            const unsigned short* g = Vb + (size_t)d * SEQ + kc + s;
            unsigned l = vbase + (unsigned)(buf * (HDIM * 32) + d * 32 + s) * 2u;
            async_copy_b128(l, g);
        }
    };

    // Q A-fragments for D=64 (two K-chunks of 32), loaded once.
    bf16x16 qa0 = load_frag_A(Qb, HDIM, qbase + row, 0, lane);
    bf16x16 qa1 = load_frag_A(Qb, HDIM, qbase + row, 32, lane);

    f32x8 O[4];
    float mrow[8], lrow[8];
#pragma unroll
    for (int nt = 0; nt < 4; ++nt)
#pragma unroll
        for (int r = 0; r < 8; ++r) O[nt][r] = 0.0f;
#pragma unroll
    for (int r = 0; r < 8; ++r) { mrow[r] = -1e30f; lrow[r] = 0.0f; }

    // prologue: stage chunk 0
    stage(0, 0);
    wait_async0();
    __syncthreads();

    int it = 0;
    for (int kc = 0; kc < SEQ; kc += 32, ++it) {
        const int cur = it & 1;
        if (kc + 32 < SEQ) stage(cur ^ 1, kc + 32);

        const unsigned short* Kl = &lds_k[cur][0];
        const unsigned short* Vl = &lds_v[cur][0];

        // ---- scores S[16 x 32] = Q(16x64) * Kchunk^T (from LDS) ----
        f32x8 s[2];
#pragma unroll
        for (int nt = 0; nt < 2; ++nt) {
            bf16x16 kb0 = load_frag_B(Kl, HDIM, nt * 16 + row, 0, lane);
            bf16x16 kb1 = load_frag_B(Kl, HDIM, nt * 16 + row, 32, lane);
            f32x8 z;
#pragma unroll
            for (int r = 0; r < 8; ++r) z[r] = 0.0f;
            z = wmma_bf16(qa0, kb0, z);
            s[nt] = wmma_bf16(qa1, kb1, z);
        }

        // ---- online softmax (per VGPR row, 16-lane DPP butterflies) ----
#pragma unroll
        for (int r = 0; r < 8; ++r) {
            float x = red16_max(fmaxf(s[0][r], s[1][r]));
            float mnew  = fmaxf(mrow[r], x);
            float scale = __expf(mrow[r] - mnew);
            mrow[r] = mnew;
            float p0 = __expf(s[0][r] - mnew);
            float p1 = __expf(s[1][r] - mnew);
            float rs = red16_sum(p0 + p1);
            lrow[r] = lrow[r] * scale + rs;
            s[0][r] = p0; s[1][r] = p1;
#pragma unroll
            for (int nt2 = 0; nt2 < 4; ++nt2) O[nt2][r] *= scale;
        }

        // ---- transpose P (C-layout -> A-layout) through per-wave LDS ----
#pragma unroll
        for (int r = 0; r < 8; ++r) {
            int rr = r + ((lane >> 4) << 3);
            lds_p[wave][rr * 32 + (lane & 15)]      = f2bf(s[0][r]);
            lds_p[wave][rr * 32 + 16 + (lane & 15)] = f2bf(s[1][r]);
        }
        FragAB pf;
        {
            const unsigned short* p = &lds_p[wave][row * 32] + ((lane >> 4) << 3);
            pf.h[0] = *reinterpret_cast<const bf16x8*>(p);
            pf.h[1] = *reinterpret_cast<const bf16x8*>(p + 16);
        }

        // ---- O += P(16x32) * Vchunk(32x64) (V from LDS) ----
#pragma unroll
        for (int nt2 = 0; nt2 < 4; ++nt2) {
            bf16x16 vb = load_frag_B(Vl, 32, nt2 * 16 + row, 0, lane);
            O[nt2] = wmma_bf16(pf.v, vb, O[nt2]);
        }

        // my async copies into the other buffer are done; everyone in sync
        wait_async0();
        __syncthreads();
    }

    // ---- finalize: O /= l, write ctx bf16 [B*T, E] ----
#pragma unroll
    for (int r = 0; r < 8; ++r) {
        float inv = 1.0f / lrow[r];
        int t = qbase + r + ((lane >> 4) << 3);
#pragma unroll
        for (int nt2 = 0; nt2 < 4; ++nt2) {
            int e = h * HDIM + nt2 * 16 + (lane & 15);
            ctx[((size_t)(b * SEQ + t)) * EMB + e] = f2bf(O[nt2][r] * inv);
        }
    }
}

// ---------------------------------------------------------------------------
// Host launch
// ---------------------------------------------------------------------------
extern "C" void kernel_launch(void* const* d_in, const int* in_sizes, int n_in,
                              void* d_out, int out_size, void* d_ws, size_t ws_size,
                              hipStream_t stream) {
    (void)in_sizes; (void)n_in; (void)out_size; (void)ws_size;

    const float* query      = (const float*)d_in[0];   // [B,T,E]
    const float* in_proj_w  = (const float*)d_in[1];   // [3E,E]
    const float* in_proj_b  = (const float*)d_in[2];   // [3E]
    const float* out_proj_w = (const float*)d_in[3];   // [E,E]
    const float* out_proj_b = (const float*)d_in[4];   // [E]

    unsigned short* ws = (unsigned short*)d_ws;
    const size_t N_Q   = (size_t)MTOT * EMB;       // 8388608
    const size_t N_WIN = (size_t)3 * EMB * EMB;    // 3145728
    const size_t N_WO  = (size_t)EMB * EMB;        // 1048576
    unsigned short* qbf    = ws;
    unsigned short* winbf  = qbf    + N_Q;
    unsigned short* woutbf = winbf  + N_WIN;
    unsigned short* q_bhtd = woutbf + N_WO;        // [BH,T,D]
    unsigned short* k_bhtd = q_bhtd + N_Q;         // [BH,T,D]
    unsigned short* v_bhdt = k_bhtd + N_Q;         // [BH,D,T]
    unsigned short* ctx    = v_bhdt + N_Q;         // [B*T,E]

    cvt_f32_bf16<<<(int)((N_Q   + 255) / 256), 256, 0, stream>>>(query,      qbf,    (int)N_Q);
    cvt_f32_bf16<<<(int)((N_WIN + 255) / 256), 256, 0, stream>>>(in_proj_w,  winbf,  (int)N_WIN);
    cvt_f32_bf16<<<(int)((N_WO  + 255) / 256), 256, 0, stream>>>(out_proj_w, woutbf, (int)N_WO);

    gemm_bf16_kernel<0><<<dim3(3 * EMB / 128, MTOT / 128), 256, 0, stream>>>(
        qbf, winbf, in_proj_b, MTOT, 3 * EMB, EMB,
        q_bhtd, k_bhtd, v_bhdt, nullptr);

    flash_attn_kernel<<<dim3(SEQ / 128, BH), 256, 0, stream>>>(
        q_bhtd, k_bhtd, v_bhdt, ctx);

    gemm_bf16_kernel<1><<<dim3(EMB / 128, MTOT / 128), 256, 0, stream>>>(
        ctx, woutbf, out_proj_b, MTOT, EMB, EMB,
        nullptr, nullptr, nullptr, (float*)d_out);
}
